// FacetAttention_53395033424314
// MI455X (gfx1250) — compile-verified, matrix operand order checked by reading, not run on hardware
//
#include <hip/hip_runtime.h>
#include <hip/hip_bf16.h>
#include <math.h>

// Problem constants (fixed by the reference setup)
constexpr int CB  = 2;     // batch
constexpr int N1  = 512;   // queries
constexpr int N2  = 512;   // keys
constexpr int DD  = 64;    // head dim
constexpr int FF  = 256;   // facet dim = 4*D
constexpr int KSTRIDE = DD + 4;  // padded LDS row stride (68 floats) to stagger banks
                                 // (matches TDM pad: 4 DWORDs after every 64 DWORDs)

typedef __bf16          v16bf __attribute__((ext_vector_type(16)));
typedef unsigned short  v16u  __attribute__((ext_vector_type(16)));
typedef float           v8f   __attribute__((ext_vector_type(8)));
typedef unsigned int    u32x4 __attribute__((ext_vector_type(4)));
typedef int             i32x4 __attribute__((ext_vector_type(4)));
typedef int             i32x8 __attribute__((ext_vector_type(8)));

static __device__ inline unsigned short f2bf(float x) {
    unsigned int u = __builtin_bit_cast(unsigned int, x);
    // round-to-nearest-even bf16
    return (unsigned short)((u + 0x7FFFu + ((u >> 16) & 1u)) >> 16);
}

// K index inside a 32-wide chunk for bf16 A/B fragment element e, lane l:
//   lanes 0-15:  elems 0-7 -> K 0-7,  elems 8-15 -> K 16-23
//   lanes 16-31: elems 0-7 -> K 8-15, elems 8-15 -> K 24-31
static __device__ inline int kmap(int e, int laneHalf) {
    return ((((e & 8) << 1) | (e & 7)) + (laneHalf << 3));
}

// ---------------------------------------------------------------------------
// TDM: issue one tensor_load_to_lds moving a [rows x 64 f32] tile from global
// memory into LDS at lds_byte_off, inserting 4 DWORDs of pad after every 64
// DWORDs (=> 68-float LDS row stride).  Descriptor packing per
// CDNA5 ISA.
// ---------------------------------------------------------------------------
static __device__ inline void tdm_load_K(const void* gaddr, unsigned lds_byte_off,
                                         unsigned rows) {
    unsigned long long ga = (unsigned long long)gaddr;
    u32x4 g0;
    g0[0] = 1u;                                   // count=1, is_restore=0, gather=0
    g0[1] = lds_byte_off;                         // LDS destination (bytes)
    g0[2] = (unsigned)(ga & 0xFFFFFFFFu);         // global addr [31:0]
    g0[3] = (unsigned)((ga >> 32) & 0x01FFFFFFu)  // global addr [56:32]
          | (2u << 30);                           // type = 2 ("image")
    i32x8 g1;
    g1[0] = (2 << 16)      // data_size = 2 -> 4 bytes
          | (1 << 20)      // pad_enable
          | (5 << 22)      // pad_interval: 5 -> every 64 DWORDs
          | (3 << 25);     // pad_amount:  3 -> 4 DWORDs
    g1[1] = (int)(64u << 16);          // tensor_dim0 = 64 (bits 79:48, low half)
    g1[2] = (int)(rows << 16);         // tensor_dim1 = rows (bits 111:80, low half)
    g1[3] = (int)(64u << 16);          // tile_dim0 = 64 (bits 127:112)
    g1[4] = (int)rows;                 // tile_dim1 = rows (bits 143:128)
    g1[5] = 64;                        // tensor_dim0_stride = 64 (bits 207:160)
    g1[6] = 0;
    g1[7] = 0;
    i32x4 z4 = {0, 0, 0, 0};
#if __has_include(<hip/amd_detail/amd_gfx1250_TDM.h>)
    i32x8 z8 = {0, 0, 0, 0, 0, 0, 0, 0};
    __builtin_amdgcn_tensor_load_to_lds(g0, g1, z4, z4, z8, 0);
#else
    __builtin_amdgcn_tensor_load_to_lds(g0, g1, z4, z4, 0);
#endif
}

// ---------------------------------------------------------------------------
// Prep kernel: convert W1 (f32 [g][f], row-major, g=out, f=in) into bf16
// fragments laid out exactly as the WMMA B-operand wants them:
//   flat index = ((gtile*8 + ktile)*32 + lane)*16 + elem
// so each lane of the GEMM loop does one contiguous 32-byte load per fragment.
// ---------------------------------------------------------------------------
__global__ __launch_bounds__(256) void w1_to_bf16_tiled(
        const float* __restrict__ W1, unsigned short* __restrict__ out) {
    int idx = blockIdx.x * 256 + threadIdx.x;
    if (idx >= FF * FF) return;
    int e  = idx & 15;
    int l  = (idx >> 4) & 31;
    int t  = (idx >> 9) & 7;     // ktile
    int G  = idx >> 12;          // gtile
    int g  = G * 16 + (l & 15);  // B-matrix column N = output unit
    int f  = t * 32 + kmap(e, l >> 4);
    out[idx] = f2bf(W1[g * FF + f]);
}

// ---------------------------------------------------------------------------
// Fused facet-attention: one workgroup per (b, i) row.
// ---------------------------------------------------------------------------
__global__ __launch_bounds__(256) void facet_attn(
        const float* __restrict__ Qg, const float* __restrict__ Kg,
        const float* __restrict__ Vg, const unsigned char* __restrict__ maskg,
        const unsigned short* __restrict__ W1t, const float* __restrict__ W2g,
        float* __restrict__ outg) {
    extern __shared__ float smem[];
    float* Ks  = smem;                   // N2 * KSTRIDE (TDM-padded rows)
    float* Qs  = Ks  + N2 * KSTRIDE;     // DD
    float* w2s = Qs  + DD;               // FF
    float* Asc = w2s + FF;               // N2 (logits, then probabilities)
    float* red = Asc + N2;               // 256 (reductions / PV partials)

    const int tid = threadIdx.x;
    const int bi  = blockIdx.x;
    const int b   = bi >> 9;             // / N1
    const int i   = bi & (N1 - 1);

    // ---- stage K rows for batch b into padded LDS via the Tensor Data Mover.
    // Wave 0 issues one 2-D tiled DMA (with hardware row padding) and waits on
    // TENSORcnt; the other waves load Q/W2 meanwhile; barrier publishes LDS.
    if (tid < 32) {
        unsigned ldsBase = __builtin_amdgcn_groupstaticsize();  // dyn LDS offset
        tdm_load_K(Kg + (size_t)b * N2 * DD, ldsBase, (unsigned)N2);
        __builtin_amdgcn_s_wait_tensorcnt(0);
    }
    if (tid < DD) Qs[tid] = Qg[((size_t)b * N1 + i) * DD + tid];
    w2s[tid] = W2g[tid];                 // FF == 256 == blockDim
    __syncthreads();

    const int lane = tid & 31;
    const int wave = tid >> 5;
    const int m    = lane & 15;          // A-matrix row within tile
    const int lh   = lane >> 4;          // lane half

    const v16u* Wt = (const v16u*)W1t;

    // ---- GEMM phase: logits A[j] = W2 . relu(W1 . facets(i,j)) ----
    for (int jt = wave; jt < N2 / 16; jt += 8) {
        const int jbase = jt * 16;
        const int j     = jbase + m;

        // Build the 8 bf16 A fragments (16 rows x 32 K) for this j-tile.
        // facet f = t*32 + k; type = f/64 = t>>1; d = f%64 = (t&1)*32 + k.
        v16bf afrag[8];
        #pragma unroll
        for (int t = 0; t < 8; ++t) {
            v16u au;
            #pragma unroll
            for (int e = 0; e < 16; ++e) {
                int k  = kmap(e, lh);
                int d  = ((t & 1) << 5) | k;
                float q  = Qs[d];
                float kv = Ks[j * KSTRIDE + d];
                int ty = t >> 1;
                float val = (ty == 0) ? q
                          : (ty == 1) ? kv
                          : (ty == 2) ? fabsf(q - kv)
                                      : q * kv;
                au[e] = f2bf(val);
            }
            afrag[t] = __builtin_bit_cast(v16bf, au);
        }

        float acc[8] = {0.f, 0.f, 0.f, 0.f, 0.f, 0.f, 0.f, 0.f};
        for (int gt = 0; gt < FF / 16; ++gt) {
            v8f c = {0.f, 0.f, 0.f, 0.f, 0.f, 0.f, 0.f, 0.f};
            #pragma unroll
            for (int t = 0; t < 8; ++t) {
                v16bf bfrag = __builtin_bit_cast(v16bf, Wt[(gt * 8 + t) * 32 + lane]);
                c = __builtin_amdgcn_wmma_f32_16x16x32_bf16(
                        /*neg_a=*/false, afrag[t],
                        /*neg_b=*/false, bfrag,
                        /*c_mod=*/(short)0, c,
                        /*reuse_a=*/false, /*reuse_b=*/false);
            }
            // C/D layout: VGPR r, lanes 0-15 -> row r, lanes 16-31 -> row r+8;
            // column g = gt*16 + (lane&15).  ReLU then dot with W2.
            float w2v = w2s[gt * 16 + m];
            #pragma unroll
            for (int r = 0; r < 8; ++r)
                acc[r] += fmaxf(c[r], 0.0f) * w2v;
        }

        // Reduce each row's partial across its 16 lanes (stay in each half).
        #pragma unroll
        for (int r = 0; r < 8; ++r) {
            float v = acc[r];
            v += __shfl_xor(v, 1, 32);
            v += __shfl_xor(v, 2, 32);
            v += __shfl_xor(v, 4, 32);
            v += __shfl_xor(v, 8, 32);
            if (m == 0) Asc[jbase + (lh << 3) + r] = v;
        }
    }
    __syncthreads();

    // ---- mask + softmax over j (each thread owns j = tid and tid+256) ----
    const unsigned char* mb = maskg + (size_t)b * N2;
    float a0 = mb[tid]       ? Asc[tid]       : -__builtin_inff();
    float a1 = mb[tid + 256] ? Asc[tid + 256] : -__builtin_inff();
    red[tid] = fmaxf(a0, a1);
    __syncthreads();
    for (int s = 128; s > 0; s >>= 1) {
        if (tid < s) red[tid] = fmaxf(red[tid], red[tid + s]);
        __syncthreads();
    }
    float mx = red[0];
    __syncthreads();
    float e0 = __expf(a0 - mx);
    float e1 = __expf(a1 - mx);
    Asc[tid] = e0; Asc[tid + 256] = e1;
    red[tid] = e0 + e1;
    __syncthreads();
    for (int s = 128; s > 0; s >>= 1) {
        if (tid < s) red[tid] += red[tid + s];
        __syncthreads();
    }
    float inv = 1.0f / red[0];
    __syncthreads();

    // ---- out[b,i,d] = (1/sum) * sum_j P[j] * V[b,j,d] ----
    const int d     = tid & 63;
    const int chunk = tid >> 6;            // 4 chunks of 128 j each
    const float* Vb = Vg + (size_t)b * N2 * DD;
    float s = 0.f;
    for (int j = chunk * 128; j < (chunk + 1) * 128; ++j)
        s += Asc[j] * Vb[j * DD + d];
    red[tid] = s;
    __syncthreads();
    if (tid < DD) {
        float o = (red[tid] + red[64 + tid] + red[128 + tid] + red[192 + tid]) * inv;
        outg[((size_t)b * N1 + i) * DD + tid] = o;
    }
}

extern "C" void kernel_launch(void* const* d_in, const int* in_sizes, int n_in,
                              void* d_out, int out_size, void* d_ws, size_t ws_size,
                              hipStream_t stream) {
    const float*         Q    = (const float*)d_in[0];
    const float*         K    = (const float*)d_in[1];
    const float*         V    = (const float*)d_in[2];
    const unsigned char* mask = (const unsigned char*)d_in[3];
    const float*         W1   = (const float*)d_in[4];
    const float*         W2   = (const float*)d_in[5];
    float*               out  = (float*)d_out;
    unsigned short*      W1t  = (unsigned short*)d_ws;   // 128 KB bf16 tiled W1

    // 1) convert W1 into WMMA-ready bf16 fragment layout (deterministic each call)
    w1_to_bf16_tiled<<<(FF * FF) / 256, 256, 0, stream>>>(W1, W1t);

    // 2) fused attention: one workgroup per (b,i)
    size_t lds = (size_t)(N2 * KSTRIDE + DD + FF + N2 + 256 + 8) * sizeof(float);
    facet_attn<<<CB * N1, 256, lds, stream>>>(Q, K, V, mask, W1t, W2, out);
}
